// DrugSynergyGNN_88510686036239
// MI455X (gfx1250) — matrix operand order repaired and checked by reading, not compiled, not run on hardware
//
#include <hip/hip_runtime.h>
#include <hip/hip_bf16.h>

// ---------------------------------------------------------------------------
// Heterogeneous GAT (3 layers, 4 edge types) for MI455X / gfx1250.
//
// Roofline: dense projections ~18 GFLOP total (trivial for the WMMA pipes);
// the limiter is ~4-5 GB of irregular edge gather/scatter at 23.3 TB/s HBM.
// Dense parts: v_wmma_f32_16x16x32_f16 fed by fragment-ordered operands
//   (B pre-swizzled to f16 in global, loaded as b128; A staged fragment-
//    ordered in 1KB LDS, fragments read as 2x ds_load_b128).
// Edge parts: wave-per-edge coalesced float4 gathers + f32 atomics.
// ---------------------------------------------------------------------------

typedef _Float16 half16 __attribute__((ext_vector_type(16)));
typedef _Float16 half8  __attribute__((ext_vector_type(8)));
typedef _Float16 half4  __attribute__((ext_vector_type(4)));
typedef float    float8 __attribute__((ext_vector_type(8)));

// ---------------------------------------------------------------------------
// Pre-swizzle weights B[K x 128] (f32, row-major) into WMMA fragment order:
// half index = (((ks*8 + w)*32 + lane)*16 + j)
//   n = w*16 + (lane & 15),  k = ks*32 + (lane>=16 ? 16 : 0) + j
// One thread per half4 group (j aligned to 4 => k run of 4, same n).
// ---------------------------------------------------------------------------
__global__ void pack_b_kernel(const float* __restrict__ B,
                              _Float16* __restrict__ Bsw, int K)
{
    int g = blockIdx.x * blockDim.x + threadIdx.x;
    int total = (K * 128) >> 2;
    if (g >= total) return;
    int j4   = (g & 3) << 2;
    int lane = (g >> 2) & 31;
    int w    = (g >> 7) & 7;
    int ks   = g >> 10;
    int n     = w * 16 + (lane & 15);
    int kbase = ks * 32 + ((lane >= 16) ? 16 : 0) + j4;
    half4 hv;
    #pragma unroll
    for (int t = 0; t < 4; ++t)
        hv[t] = (_Float16)B[(size_t)(kbase + t) * 128 + n];
    *(half4*)(Bsw + (size_t)g * 4) = hv;
}

// ---------------------------------------------------------------------------
// C[M x 128] = act( (A*ascale)[M x K] * B[K x 128] + bias )
// Block: 256 threads = 8 waves; block owns 16 rows, wave w owns cols [16w,16w+16).
// Bsw: fragment-ordered f16 weights from pack_b_kernel. K % 32 == 0, M % 16 == 0.
// ---------------------------------------------------------------------------
__global__ __launch_bounds__(256) void gemm_bias_act_kernel(
    const float* __restrict__ A, const _Float16* __restrict__ Bsw,
    const float* __restrict__ bias, float* __restrict__ C,
    int M, int K, float ascale, int act)
{
    __shared__ _Float16 lA[32 * 16];   // fragment-ordered A tile (1 KB)

    const int tid  = threadIdx.x;
    const int lane = tid & 31;
    const int wv   = tid >> 5;          // 0..7
    const int m0   = blockIdx.x * 16;
    const int l15  = lane & 15;
    const int hi   = (lane >= 16);

    // A staging role (threads 0..127): float4 run along K -> one half4 slot
    const int sr  = tid >> 3;           // row 0..15
    const int sk4 = (tid & 7) << 2;     // k offset 0,4,...,28
    const int slane = sr + (((sk4 >> 4) & 1) << 4);
    const int sj    = sk4 & 15;
    _Float16* sdst = lA + slane * 16 + sj;

    float8 c = {};

    for (int k0 = 0; k0 < K; k0 += 32) {
        if (tid < 128) {
            const float4 v = *(const float4*)(A + (size_t)(m0 + sr) * K + (k0 + sk4));
            half4 hv;
            hv[0] = (_Float16)(v.x * ascale);
            hv[1] = (_Float16)(v.y * ascale);
            hv[2] = (_Float16)(v.z * ascale);
            hv[3] = (_Float16)(v.w * ascale);
            *(half4*)sdst = hv;
        }
        if (k0 + 32 < K)   // hint next A slab (global_prefetch_b8)
            __builtin_prefetch(A + (size_t)(m0 + sr) * K + (k0 + 32), 0, 1);
        __syncthreads();

        // A fragment: 2x ds_load_b128
        half8 a0 = *(const half8*)(lA + lane * 16);
        half8 a1 = *(const half8*)(lA + lane * 16 + 8);
        half16 af = __builtin_shufflevector(a0, a1, 0, 1, 2, 3, 4, 5, 6, 7,
                                            8, 9, 10, 11, 12, 13, 14, 15);
        // B fragment: 2x global_load_b128 from the swizzled weights
        const _Float16* bp = Bsw + (((size_t)(k0 >> 5) * 8 + wv) * 512) + lane * 16;
        half8 b0 = *(const half8*)(bp);
        half8 b1 = *(const half8*)(bp + 8);
        half16 bf = __builtin_shufflevector(b0, b1, 0, 1, 2, 3, 4, 5, 6, 7,
                                            8, 9, 10, 11, 12, 13, 14, 15);

        c = __builtin_amdgcn_wmma_f32_16x16x32_f16(
                false, af, false, bf, (short)0, c, false, false);
        __syncthreads();
    }

    // epilogue: C/D layout: col = lane&15 (+16w), row r -> m0 + r + (hi?8:0)
    const int col = wv * 16 + l15;
    const float bv = bias ? bias[col] : 0.0f;
    #pragma unroll
    for (int r = 0; r < 8; ++r) {
        int m = m0 + r + (hi ? 8 : 0);
        float v = c[r] + bv;
        if (act) v = fmaxf(v, 0.0f);
        C[(size_t)m * 128 + col] = v;
    }
}

// ---------------------------------------------------------------------------
// el[n,h] = sum_d z[n, h*32+d] * al[h,d]      (thread per node)
// ---------------------------------------------------------------------------
__global__ void compute_el_kernel(const float* __restrict__ z,
                                  const float* __restrict__ al,
                                  float* __restrict__ el, int n)
{
    int i = blockIdx.x * blockDim.x + threadIdx.x;
    if (i >= n) return;
    const float* row = z + (size_t)i * 128;
    #pragma unroll
    for (int h = 0; h < 4; ++h) {
        float s = 0.f;
        #pragma unroll 8
        for (int d = 0; d < 32; ++d) s += row[h * 32 + d] * al[h * 32 + d];
        el[i * 4 + h] = s;
    }
}

// wr[k,h] = sum_d W[k, h*32+d] * ar[h,d]   (512 outputs)
__global__ void compute_wr_kernel(const float* __restrict__ W,
                                  const float* __restrict__ ar,
                                  float* __restrict__ wr)
{
    int idx = blockIdx.x * blockDim.x + threadIdx.x;
    if (idx >= 512) return;
    int k = idx >> 2, h = idx & 3;
    float s = 0.f;
    #pragma unroll 8
    for (int d = 0; d < 32; ++d) s += W[k * 128 + h * 32 + d] * ar[h * 32 + d];
    wr[k * 4 + h] = s;
}

// er[n,h] = sum_k hp[n,k] * wr[k,h]
__global__ void compute_er_kernel(const float* __restrict__ hp,
                                  const float* __restrict__ wr,
                                  float* __restrict__ er, int n)
{
    __shared__ float swr[512];
    for (int i = threadIdx.x; i < 512; i += blockDim.x) swr[i] = wr[i];
    __syncthreads();
    int i = blockIdx.x * blockDim.x + threadIdx.x;
    if (i >= n) return;
    const float* row = hp + (size_t)i * 128;
    float a0 = 0.f, a1 = 0.f, a2 = 0.f, a3 = 0.f;
    for (int k = 0; k < 128; ++k) {
        float v = row[k];
        a0 += v * swr[k * 4 + 0]; a1 += v * swr[k * 4 + 1];
        a2 += v * swr[k * 4 + 2]; a3 += v * swr[k * 4 + 3];
    }
    er[i * 4 + 0] = a0; er[i * 4 + 1] = a1;
    er[i * 4 + 2] = a2; er[i * 4 + 3] = a3;
}

__global__ void fill_f32_kernel(float* __restrict__ p, float v, int n)
{
    int i = blockIdx.x * blockDim.x + threadIdx.x;
    if (i < n) p[i] = v;
}

// acc row-broadcast init with summed GAT biases
__global__ void acc_init_kernel(float* __restrict__ acc,
                                const float* __restrict__ b0,
                                const float* __restrict__ b1, int total)
{
    int i = blockIdx.x * blockDim.x + threadIdx.x;
    if (i >= total) return;
    int c = i & 127;
    float v = b0[c];
    if (b1) v += b1[c];
    acc[i] = v;
}

__device__ inline void atomicMaxF32(float* addr, float val)
{
    // sign-aware bit trick: works for any mix of pos/neg values
    if (val >= 0.0f) atomicMax((int*)addr, __float_as_int(val));
    else             atomicMin((unsigned int*)addr, __float_as_uint(val));
}

// pass 1: segment max of leaky_relu(el[src]+er[dst]) per (dst, head)
__global__ void edge_max_kernel(const int* __restrict__ src, const int* __restrict__ dst,
                                const float* __restrict__ el, const float* __restrict__ er,
                                float* __restrict__ emax, int E)
{
    int t = blockIdx.x * blockDim.x + threadIdx.x;
    if (t >= E * 4) return;
    int e = t >> 2, h = t & 3;
    int s = src[e], d = dst[e];
    float v = el[s * 4 + h] + er[d * 4 + h];
    v = (v >= 0.f) ? v : 0.2f * v;
    atomicMaxF32(&emax[d * 4 + h], v);
}

// pass 2: ee = exp(v - emax[dst]); denom[dst] += ee
__global__ void edge_exp_sum_kernel(const int* __restrict__ src, const int* __restrict__ dst,
                                    const float* __restrict__ el, const float* __restrict__ er,
                                    const float* __restrict__ emax,
                                    float* __restrict__ ee, float* __restrict__ denom, int E)
{
    int t = blockIdx.x * blockDim.x + threadIdx.x;
    if (t >= E * 4) return;
    int e = t >> 2, h = t & 3;
    int s = src[e], d = dst[e];
    float v = el[s * 4 + h] + er[d * 4 + h];
    v = (v >= 0.f) ? v : 0.2f * v;
    float x = __expf(v - emax[d * 4 + h]);
    ee[t] = x;
    atomicAdd(&denom[d * 4 + h], x);
}

// pass 3: one wave per edge; lane l handles channels [4l, 4l+3] (head = l>>3)
__global__ void edge_message_kernel(const int* __restrict__ src, const int* __restrict__ dst,
                                    const float* __restrict__ ee, const float* __restrict__ denom,
                                    const float* __restrict__ z, float* __restrict__ acc, int E)
{
    int gw = (int)((blockIdx.x * blockDim.x + threadIdx.x) >> 5);
    int lane = threadIdx.x & 31;
    if (gw >= E) return;
    int s = src[gw], d = dst[gw];
    int h = lane >> 3;
    float alpha = ee[gw * 4 + h] / fmaxf(denom[d * 4 + h], 1e-9f);
    const float4 zv = ((const float4*)(z + (size_t)s * 128))[lane];
    float* out = acc + (size_t)d * 128 + lane * 4;
    atomicAdd(out + 0, alpha * zv.x);
    atomicAdd(out + 1, alpha * zv.y);
    atomicAdd(out + 2, alpha * zv.z);
    atomicAdd(out + 3, alpha * zv.w);
}

// ---------------------------------------------------------------------------
// Host orchestration
// ---------------------------------------------------------------------------
extern "C" void kernel_launch(void* const* d_in, const int* in_sizes, int n_in,
                              void* d_out, int out_size, void* d_ws, size_t ws_size,
                              hipStream_t stream)
{
    (void)n_in; (void)out_size; (void)ws_size;

    // ntype order internal: 0=drug, 1=target, 2=disease
    const int Nnt[3]   = {20000, 20000, 10000};
    const int K0[3]    = {512, 320, 256};
    const float* feat[3] = {(const float*)d_in[0], (const float*)d_in[1], (const float*)d_in[2]};

    // edge types: param-index (sorted order), src/dst d_in idx, src/dst ntype
    struct Et { int p, si, di, s, d; };
    const Et ets[4] = {
        {3, 3, 4,  0, 1},   // 'targets'         drug -> target
        {2, 5, 6,  1, 0},   // 'targeted_by'     target -> drug
        {0, 7, 8,  0, 2},   // 'associated_with' drug -> disease
        {1, 9, 10, 2, 1},   // 'has_target'      disease -> target
    };

    // workspace carve-up (floats)
    float* ws = (float*)d_ws;
    size_t off = 0;
    auto alloc = [&](size_t n) { float* p = ws + off; off += n; return p; };
    float *hbuf[3], *hpbuf[3], *accbuf[3];
    for (int nt = 0; nt < 3; ++nt) hbuf[nt]  = alloc((size_t)Nnt[nt] * 128);
    for (int nt = 0; nt < 3; ++nt) hpbuf[nt] = alloc((size_t)Nnt[nt] * 128);
    for (int nt = 0; nt < 3; ++nt) accbuf[nt]= alloc((size_t)Nnt[nt] * 128);
    float* zsrc  = alloc((size_t)20000 * 128);
    float* el    = alloc((size_t)20000 * 4);
    float* er    = alloc((size_t)20000 * 4);
    float* emax  = alloc((size_t)20000 * 4);
    float* denom = alloc((size_t)20000 * 4);
    float* ee    = alloc((size_t)400000 * 4);
    float* wr    = alloc(512);
    _Float16* bsw = (_Float16*)alloc(32768);   // 512*128 f16 swizzled weights

    // params: pytree flatten order (dicts sorted by key) => per layer:
    //   gat{associated_with,has_target,targeted_by,targets} x {W,al,ar,b}  (16)
    //   in_proj{disease,drug,target} x (W,b)                               (6)
    //   out_proj{disease,drug,target} x (W,b)                              (6)
    auto P = [&](int l, int j) { return (const float*)d_in[11 + 28 * l + j]; };
    const int inW[3]  = {18, 20, 16};   // drug, target, disease
    const int outW[3] = {24, 26, 22};
    const float cnt[3] = {1.f, 2.f, 1.f};  // # edge-type contributions per dst ntype

    auto gemm = [&](const float* A, const float* B, const float* bias, float* C,
                    int M, int K, float ascale, int act) {
        pack_b_kernel<<<(K * 128 / 4 + 255) / 256, 256, 0, stream>>>(B, bsw, K);
        gemm_bias_act_kernel<<<M / 16, 256, 0, stream>>>(A, bsw, bias, C, M, K, ascale, act);
    };

    for (int l = 0; l < 3; ++l) {
        // 1) in_proj + ReLU
        for (int nt = 0; nt < 3; ++nt) {
            const float* A = (l == 0) ? feat[nt] : hbuf[nt];
            int K = (l == 0) ? K0[nt] : 128;
            gemm(A, P(l, inW[nt]), P(l, inW[nt] + 1), hpbuf[nt], Nnt[nt], K, 1.0f, 1);
        }

        // 2) init accumulators with summed GAT biases (so mean = (sum+Σb)/cnt)
        {
            int td = Nnt[0] * 128;
            acc_init_kernel<<<(td + 255) / 256, 256, 0, stream>>>(
                accbuf[0], P(l, 4 * 2 + 3), (const float*)nullptr, td);   // drug <- targeted_by.b
            int tt = Nnt[1] * 128;
            acc_init_kernel<<<(tt + 255) / 256, 256, 0, stream>>>(
                accbuf[1], P(l, 4 * 3 + 3), P(l, 4 * 1 + 3), tt);         // target <- targets.b + has_target.b
            int ts = Nnt[2] * 128;
            acc_init_kernel<<<(ts + 255) / 256, 256, 0, stream>>>(
                accbuf[2], P(l, 4 * 0 + 3), (const float*)nullptr, ts);   // disease <- associated_with.b
        }

        // 3) per edge type: GAT conv
        for (int q = 0; q < 4; ++q) {
            const Et& et = ets[q];
            const float* W  = P(l, 4 * et.p + 0);
            const float* al = P(l, 4 * et.p + 1);
            const float* ar = P(l, 4 * et.p + 2);
            const int* src = (const int*)d_in[et.si];
            const int* dst = (const int*)d_in[et.di];
            int E  = in_sizes[et.si];
            int Ns = Nnt[et.s], Nd = Nnt[et.d];

            // z_src = hp[src_nt] @ W   (WMMA)
            gemm(hpbuf[et.s], W, (const float*)nullptr, zsrc, Ns, 128, 1.0f, 0);
            // attention logits
            compute_el_kernel<<<(Ns + 255) / 256, 256, 0, stream>>>(zsrc, al, el, Ns);
            compute_wr_kernel<<<2, 256, 0, stream>>>(W, ar, wr);
            compute_er_kernel<<<(Nd + 255) / 256, 256, 0, stream>>>(hpbuf[et.d], wr, er, Nd);
            // edge softmax
            fill_f32_kernel<<<(Nd * 4 + 255) / 256, 256, 0, stream>>>(emax, -3.402823466e38f, Nd * 4);
            fill_f32_kernel<<<(Nd * 4 + 255) / 256, 256, 0, stream>>>(denom, 0.0f, Nd * 4);
            edge_max_kernel<<<(E * 4 + 255) / 256, 256, 0, stream>>>(src, dst, el, er, emax, E);
            edge_exp_sum_kernel<<<(E * 4 + 255) / 256, 256, 0, stream>>>(src, dst, el, er, emax, ee, denom, E);
            // weighted scatter: acc[dst] += alpha * z_src[src]
            edge_message_kernel<<<(E + 7) / 8, 256, 0, stream>>>(src, dst, ee, denom, zsrc, accbuf[et.d], E);
        }

        // 4) out_proj(mean(acc)) (+ ReLU except last layer); last layer -> d_out
        for (int nt = 0; nt < 3; ++nt) {
            float* out;
            if (l == 2) {
                size_t o = (nt == 0) ? 0 : (nt == 1) ? (size_t)20000 * 128 : (size_t)40000 * 128;
                out = (float*)d_out + o;
            } else {
                out = hbuf[nt];
            }
            gemm(accbuf[nt], P(l, outW[nt]), P(l, outW[nt] + 1), out,
                 Nnt[nt], 128, 1.0f / cnt[nt], (l < 2) ? 1 : 0);
        }
    }
}
